// MaxViTAttn_37598143709912
// MI455X (gfx1250) — matrix operand order repaired and checked by reading, not compiled, not run on hardware
//
#include <hip/hip_runtime.h>
#include <hip/hip_bf16.h>

typedef __attribute__((ext_vector_type(16))) _Float16 v16h;
typedef __attribute__((ext_vector_type(8)))  _Float16 v8h;
typedef __attribute__((ext_vector_type(8)))  float    v8f;

#define NW   49      // tokens per 7x7 window
#define NP   64      // padded tokens (4 M-tiles)
#define CH   128     // channels
#define QKN  256     // Q+K columns
#define LN_EPS 1e-5f

static __device__ __forceinline__ v8f wmma_f16(v16h a, v16h b, v8f c) {
  // D = A(16x32 f16) * B(32x16 f16) + C(16x16 f32)
  return __builtin_amdgcn_wmma_f32_16x16x32_f16(false, a, false, b, (short)0, c,
                                                false, false);
}

// Load a 16x32 f16 A-fragment from row-major f16 LDS tile.
// Layout: lane r=lane&15 is row M; hf=lane>>4;
// elements 0..15 -> K = k0 + 16*hf + e (contiguous)
static __device__ __forceinline__ v16h load_a_frag(const _Float16* base, int stride,
                                                   int k0, int lane) {
  const int r  = lane & 15;
  const int hf = lane >> 4;
  const _Float16* p = base + r * stride + k0 + 16 * hf;
  v8h lo = *(const v8h*)p;
  v8h hi = *(const v8h*)(p + 8);
  v16h o;
#pragma unroll
  for (int i = 0; i < 8; ++i) { o[i] = lo[i]; o[i + 8] = hi[i]; }
  return o;
}

// Load 16 contiguous f16 (B fragment: col = lane&15, K = k-range split by half)
static __device__ __forceinline__ v16h load_b16(const _Float16* p) {
  v8h lo = *(const v8h*)p;
  v8h hi = *(const v8h*)(p + 8);
  v16h o;
#pragma unroll
  for (int i = 0; i < 8; ++i) { o[i] = lo[i]; o[i + 8] = hi[i]; }
  return o;
}

// ---- pre-pass: convert weights f32 -> f16 once (weights are reused by 4096 blocks)
__global__ __launch_bounds__(256)
void cvt_weights_kernel(const float* __restrict__ qkv_w,
                        const float* __restrict__ proj_w,
                        _Float16* __restrict__ wq, _Float16* __restrict__ wp) {
  const int i = blockIdx.x * 256 + threadIdx.x;
  if (i < 3 * CH * CH) wq[i] = (_Float16)qkv_w[i];
  if (i < CH * CH)     wp[i] = (_Float16)proj_w[i];
}

__global__ __launch_bounds__(256)
void maxvit_block_attn_kernel(const float* __restrict__ x,
                              const float* __restrict__ norm_w,
                              const float* __restrict__ norm_b,
                              const _Float16* __restrict__ wq16,
                              const float* __restrict__ qkv_b,
                              const _Float16* __restrict__ wp16,
                              const float* __restrict__ proj_b,
                              const float* __restrict__ pos_table,
                              float* __restrict__ out) {
  extern __shared__ char smem[];
  _Float16* lds_a  = (_Float16*)(smem);            // 64x128 f16 (LN out)      16KB
  _Float16* lds_qk = (_Float16*)(smem + 16384);    // 64x256 f16 (Q|K)         32KB
  _Float16* lds_vt = (_Float16*)(smem + 49152);    // 128x64 f16 (V^T: [c][n]) 16KB
  _Float16* lds_o  = (_Float16*)(smem + 65536);    // 64x128 f16 (attn out)    16KB
  _Float16* lds_p  = (_Float16*)(smem + 81920);    // 8 waves x 16x64 f16      16KB
  float*    lds_bias = (float*)(smem + 98304);     // 169 f32 rel-pos bias      1KB

  const int tid  = threadIdx.x;
  const int wid  = tid >> 5;
  const int lane = tid & 31;
  const int ln16 = lane & 15;
  const int hf   = lane >> 4;

  const int win = blockIdx.x;       // 4096 windows: b*256 + wy*16 + wx
  const int wx  = win & 15;
  const int wy  = (win >> 4) & 15;
  const int bb  = win >> 8;

  if (tid < 169) lds_bias[tid] = pos_table[tid];

  // ---------------- Phase 1: LayerNorm -> lds_a (f16), pad rows zeroed ------
  for (int m = wid; m < NP; m += 8) {
    _Float16* dst = lds_a + m * CH + lane * 4;
    if (m < NW) {
      const int iy = m / 7, ix = m - iy * 7;
      const long grow = (long)bb * 12544 + (long)((wy * 7 + iy) * 112 + (wx * 7 + ix));
      float4 v = ((const float4*)(x + grow * CH))[lane];
      float s  = v.x + v.y + v.z + v.w;
      float ss = v.x * v.x + v.y * v.y + v.z * v.z + v.w * v.w;
#pragma unroll
      for (int msk = 16; msk >= 1; msk >>= 1) {
        s  += __shfl_xor(s,  msk, 32);
        ss += __shfl_xor(ss, msk, 32);
      }
      const float mean = s * (1.0f / CH);
      const float var  = ss * (1.0f / CH) - mean * mean;
      const float rinv = rsqrtf(var + LN_EPS);
      float4 w4 = ((const float4*)norm_w)[lane];
      float4 b4 = ((const float4*)norm_b)[lane];
      dst[0] = (_Float16)((v.x - mean) * rinv * w4.x + b4.x);
      dst[1] = (_Float16)((v.y - mean) * rinv * w4.y + b4.y);
      dst[2] = (_Float16)((v.z - mean) * rinv * w4.z + b4.z);
      dst[3] = (_Float16)((v.w - mean) * rinv * w4.w + b4.w);
    } else {
      dst[0] = (_Float16)0.0f; dst[1] = (_Float16)0.0f;
      dst[2] = (_Float16)0.0f; dst[3] = (_Float16)0.0f;
    }
  }
  __syncthreads();

  // ---------------- Phase 2: QKV GEMM  M=64 N=384 K=128 ---------------------
  // wave w owns N-tiles jt = 3w .. 3w+2 ; B[k][j] = qkv_w[j][k] (f16 pre-pass)
  // jt 0..7 -> Q, 8..15 -> K (stored [n][c] in lds_qk); jt 16..23 -> V stored
  // transposed into lds_vt[c][n] so P@V B-fragments are contiguous.
#pragma unroll
  for (int j3 = 0; j3 < 3; ++j3) {
    const int jt = wid * 3 + j3;
    v8f acc[4];
#pragma unroll
    for (int mt = 0; mt < 4; ++mt)
#pragma unroll
      for (int i = 0; i < 8; ++i) acc[mt][i] = 0.0f;
#pragma unroll
    for (int kk = 0; kk < 4; ++kk) {
      const int j = jt * 16 + ln16;                     // output column
      v16h bf = load_b16(wq16 + j * CH + kk * 32 + hf * 16);
#pragma unroll
      for (int mt = 0; mt < 4; ++mt)
        acc[mt] = wmma_f16(load_a_frag(lds_a + mt * 16 * CH, CH, kk * 32, lane),
                           bf, acc[mt]);
    }
    const float qb = qkv_b[jt * 16 + ln16];
    if (jt < 16) {          // Q or K -> [n][c]
#pragma unroll
      for (int mt = 0; mt < 4; ++mt)
#pragma unroll
        for (int r = 0; r < 8; ++r)
          lds_qk[(mt * 16 + r + 8 * hf) * QKN + jt * 16 + ln16] =
              (_Float16)(acc[mt][r] + qb);
    } else {                // V -> transposed [c][n]
      const int c = jt * 16 + ln16 - 256;
#pragma unroll
      for (int mt = 0; mt < 4; ++mt)
#pragma unroll
        for (int r = 0; r < 8; ++r)
          lds_vt[c * NP + mt * 16 + r + 8 * hf] = (_Float16)(acc[mt][r] + qb);
    }
  }
  __syncthreads();

  // ---------------- Phase 3: attention, wave = head -------------------------
  {
    const int h = wid;
    const _Float16* Qb = lds_qk + h * 16;           // [n][d] stride QKN
    const _Float16* Kb = lds_qk + 128 + h * 16;
    _Float16* Pb = lds_p + wid * 16 * NP;           // per-wave 16x64 scratch

    // K^T fragments (B: col = key token, K = d padded 16->32): mt-invariant
    v16h kf[4];
#pragma unroll
    for (int nt = 0; nt < 4; ++nt) {
      const _Float16* p = Kb + (nt * 16 + ln16) * QKN;
      v8h lo = *(const v8h*)p;
      v8h hi = *(const v8h*)(p + 8);
#pragma unroll
      for (int i = 0; i < 8; ++i) {
        kf[nt][i]     = hf ? (_Float16)0.0f : lo[i];
        kf[nt][i + 8] = hf ? (_Float16)0.0f : hi[i];
      }
    }
    // V fragments (B: col = d, K = key token; contiguous thanks to V^T): mt-invariant
    v16h vf[2];
#pragma unroll
    for (int kk = 0; kk < 2; ++kk) {
      const _Float16* p = lds_vt + (h * 16 + ln16) * NP + kk * 32 + hf * 16;
      v8h lo = *(const v8h*)p;
      v8h hi = *(const v8h*)(p + 8);
#pragma unroll
      for (int i = 0; i < 8; ++i) { vf[kk][i] = lo[i]; vf[kk][i + 8] = hi[i]; }
    }

#pragma unroll
    for (int mt = 0; mt < 4; ++mt) {
      // Q fragment: dh=16 padded to K=32 (upper half zero)
      v16h qf;
      {
        const _Float16* p = Qb + (mt * 16 + ln16) * QKN;
        v8h lo = *(const v8h*)p;
        v8h hi = *(const v8h*)(p + 8);
#pragma unroll
        for (int i = 0; i < 8; ++i) { qf[i] = hf ? (_Float16)0.0f : lo[i]; qf[i + 8] = hf ? (_Float16)0.0f : hi[i]; }
      }
      v8f sacc[4];
#pragma unroll
      for (int nt = 0; nt < 4; ++nt) {
#pragma unroll
        for (int i = 0; i < 8; ++i) sacc[nt][i] = 0.0f;
        sacc[nt] = wmma_f16(qf, kf[nt], sacc[nt]);
      }
      // softmax over 64 keys (mask >=49); row reduction across 16-lane group
#pragma unroll
      for (int r = 0; r < 8; ++r) {
        const int row = mt * 16 + r + 8 * hf;   // query token
        float sv[4];
        float mx = -1e30f;
#pragma unroll
        for (int nt = 0; nt < 4; ++nt) {
          const int col = nt * 16 + ln16;       // key token
          float s = sacc[nt][r] * 0.25f;        // scale = dh^-0.5 = 0.25
          const int n  = (row < NW) ? row : (NW - 1);
          const int mm = (col < NW) ? col : (NW - 1);
          const int ny = n / 7,  nx = n - ny * 7;
          const int my = mm / 7, mxc = mm - my * 7;
          s += lds_bias[(ny - my + 6) * 13 + (nx - mxc + 6)];
          if (col >= NW) s = -1e30f;
          sv[nt] = s;
          mx = fmaxf(mx, s);
        }
#pragma unroll
        for (int msk = 1; msk <= 8; msk <<= 1) mx = fmaxf(mx, __shfl_xor(mx, msk, 32));
        float sum = 0.0f;
#pragma unroll
        for (int nt = 0; nt < 4; ++nt) { sv[nt] = __expf(sv[nt] - mx); sum += sv[nt]; }
#pragma unroll
        for (int msk = 1; msk <= 8; msk <<= 1) sum += __shfl_xor(sum, msk, 32);
        const float rs = 1.0f / sum;
#pragma unroll
        for (int nt = 0; nt < 4; ++nt)
          Pb[(r + 8 * hf) * NP + nt * 16 + ln16] = (_Float16)(sv[nt] * rs);
      }
      asm volatile("s_wait_dscnt 0x0" ::: "memory");  // cross-lane P in LDS ready

      // O tile = P(16x64) * V(64x16): 2 K-steps of 32
      v8f oacc;
#pragma unroll
      for (int i = 0; i < 8; ++i) oacc[i] = 0.0f;
#pragma unroll
      for (int kk = 0; kk < 2; ++kk)
        oacc = wmma_f16(load_a_frag(Pb, NP, kk * 32, lane), vf[kk], oacc);
#pragma unroll
      for (int r = 0; r < 8; ++r)
        lds_o[(mt * 16 + r + 8 * hf) * CH + h * 16 + ln16] = (_Float16)oacc[r];
    }
  }
  __syncthreads();

  // ---------------- Phase 4: proj GEMM + bias + residual --------------------
  {
    const int nt = wid;                               // wave owns one 16-col tile
    v8f acc[4];
#pragma unroll
    for (int mt = 0; mt < 4; ++mt)
#pragma unroll
      for (int i = 0; i < 8; ++i) acc[mt][i] = 0.0f;
#pragma unroll
    for (int kk = 0; kk < 4; ++kk) {
      const int j = nt * 16 + ln16;
      v16h bf = load_b16(wp16 + j * CH + kk * 32 + hf * 16);
#pragma unroll
      for (int mt = 0; mt < 4; ++mt)
        acc[mt] = wmma_f16(load_a_frag(lds_o + mt * 16 * CH, CH, kk * 32, lane),
                           bf, acc[mt]);
    }
    const float pb = proj_b[nt * 16 + ln16];
#pragma unroll
    for (int mt = 0; mt < 4; ++mt) {
#pragma unroll
      for (int r = 0; r < 8; ++r) {
        const int row = mt * 16 + r + 8 * hf;
        if (row < NW) {
          const int iy = row / 7, ix = row - iy * 7;
          const long grow =
              (long)bb * 12544 + (long)((wy * 7 + iy) * 112 + (wx * 7 + ix));
          const long idx = grow * CH + nt * 16 + ln16;
          out[idx] = x[idx] + acc[mt][r] + pb;
        }
      }
    }
  }
}

extern "C" void kernel_launch(void* const* d_in, const int* in_sizes, int n_in,
                              void* d_out, int out_size, void* d_ws, size_t ws_size,
                              hipStream_t stream) {
  const float* x         = (const float*)d_in[0];
  const float* norm_w    = (const float*)d_in[1];
  const float* norm_b    = (const float*)d_in[2];
  const float* qkv_w     = (const float*)d_in[3];
  const float* qkv_b     = (const float*)d_in[4];
  const float* proj_w    = (const float*)d_in[5];
  const float* proj_b    = (const float*)d_in[6];
  const float* pos_table = (const float*)d_in[7];
  float* out = (float*)d_out;

  // workspace: f16 copies of qkv_w (98304 B) and proj_w (32768 B)
  _Float16* wq16 = (_Float16*)d_ws;
  _Float16* wp16 = wq16 + 3 * CH * CH;

  hipLaunchKernelGGL(cvt_weights_kernel, dim3(192), dim3(256), 0, stream,
                     qkv_w, proj_w, wq16, wp16);

  const size_t shmem = 16384 + 32768 + 16384 + 16384 + 16384 + 1024;  // 99328 B
  hipLaunchKernelGGL(maxvit_block_attn_kernel, dim3(4096), dim3(256), shmem, stream,
                     x, norm_w, norm_b, wq16, qkv_b, wp16, proj_b, pos_table, out);
}
